// GCN_layer_27376121545349
// MI455X (gfx1250) — matrix operand order repaired and checked by reading, not compiled, not run on hardware
//
#include <hip/hip_runtime.h>
#include <stdint.h>

// ---------------------------------------------------------------------------
// GCN layer for MI455X (gfx1250, wave32, WMMA).
//   h  = relu(x @ W1i + b1i) @ W2i + b2i          (bf16 WMMA, f32 accum)
//   ha = segment_sum(h[src], dst)                  (bf16 gather, f32 atomics)
//   y  = relu(ha @ W1f + b1f) @ W2f + b2f          (bf16 WMMA, f32 accum)
// ---------------------------------------------------------------------------

typedef __bf16 bf16;
typedef __attribute__((ext_vector_type(16))) __bf16 v16bf;
typedef __attribute__((ext_vector_type(8)))  __bf16 v8bf;
typedef __attribute__((ext_vector_type(4)))  __bf16 v4bf;
typedef __attribute__((ext_vector_type(8)))  float  v8f;

#define N_NODES 50000
#define N_EDGES 800000
#define IN_DIM  128
#define HI_DIM  512
#define EM_DIM  256
#define OUT_DIM 128

// ---------------------------------------------------------------------------
// Pack W [K,N] f32 (row-major) into bf16 B-fragment records:
//   pk[((kt*(N/16)+nt)*32 + lane)*16 + j]  =  W[kt*32 + (lane<16?0:16) + j][nt*16 + lane%16]
// so a wave loads its 32x16 B fragment with one contiguous 32B read per lane.
// ---------------------------------------------------------------------------
__global__ __launch_bounds__(256) void pack_w_kernel(const float* __restrict__ W,
                                                     bf16* __restrict__ pk,
                                                     int K, int N) {
    int idx   = blockIdx.x * 256 + threadIdx.x;      // one packed lane-record
    int total = (K >> 5) * (N >> 4) * 32;
    if (idx >= total) return;
    int l      = idx & 31;
    int t      = idx >> 5;
    int ntiles = N >> 4;
    int nt     = t % ntiles;
    int kt     = t / ntiles;
    int kbase  = (kt << 5) + ((l >> 4) << 4);        // +0 or +16
    int n      = (nt << 4) + (l & 15);
    bf16* dst  = pk + (size_t)idx * 16;
#pragma unroll
    for (int j = 0; j < 16; ++j)
        dst[j] = (bf16)W[(size_t)(kbase + j) * N + n];
}

// ---------------------------------------------------------------------------
// load 4 consecutive elements, converting to bf16 (vector path for staging)
// ---------------------------------------------------------------------------
__device__ inline v4bf load4_cvt(const float* p) {
    const float4 f = *(const float4*)p;
    v4bf r;
    r[0] = (bf16)f.x; r[1] = (bf16)f.y; r[2] = (bf16)f.z; r[3] = (bf16)f.w;
    return r;
}
__device__ inline v4bf load4_cvt(const bf16* p) {
    return *(const v4bf*)p;
}

// ---------------------------------------------------------------------------
// WMMA GEMM: C[M,N] = act(A[M,K] @ W[K,N] + bias), M = N_NODES
//   block = 256 threads = 8 waves; block tile = 32 rows x 128 cols.
//   Each wave owns one 16-col N-tile and TWO 16-row accumulators sharing one
//   B fragment -> 2 WMMAs per barrier, B fragment loaded once per k-step.
//   A tile (32x32) double-buffered in LDS: stage k+1 while computing k, so
//   only ONE __syncthreads per k-step.
// ---------------------------------------------------------------------------
template <typename TIN, typename TOUT, bool RELU, int K, int N>
__global__ __launch_bounds__(256) void gemm_bias_act(const TIN* __restrict__ A,
                                                     const bf16* __restrict__ pkB,
                                                     const float* __restrict__ bias,
                                                     TOUT* __restrict__ C) {
    constexpr int LDSS  = 40;                        // padded row stride (bank spread)
    constexpr int BUFSZ = 32 * LDSS;
    __shared__ __align__(16) bf16 As[2 * BUFSZ];

    const int tid   = threadIdx.x;
    const int lane  = tid & 31;
    const int wave  = tid >> 5;
    const int rbase = blockIdx.y << 5;               // 32 rows per block
    const int ntile = (blockIdx.x << 3) + wave;      // 8 N-tiles per block
    const int col   = (ntile << 4) + (lane & 15);

    const int hi    = lane >> 4;                     // 0 or 1 (half-wave)
    const int arow  = lane & 15;
    const int aoff0 = hi ? 16 : 0;                   // contiguous k = 16*hi + s (matches packed B)
    const int aoff1 = hi ? 24 : 8;

    // staging coords: each thread stages 4 consecutive elements of one row
    const int srow  = tid >> 3;                      // 0..31
    const int scol  = (tid & 7) << 2;                // 0,4,...,28
    int  grow  = rbase + srow;
    if (grow >= N_NODES) grow = N_NODES - 1;         // clamp (writeback predicated)
    const TIN* ap = &A[(size_t)grow * K + scol];

    v8f acc0 = {};
    v8f acc1 = {};

    const bf16* bp = pkB + ((size_t)ntile * 32 + lane) * 16;
    constexpr size_t bstep = (size_t)(N >> 4) * 32 * 16;   // elems per k-step
    constexpr int KT = K >> 5;

    // prologue: stage k-step 0 into buffer 0
    *(v4bf*)&As[srow * LDSS + scol] = load4_cvt(ap);

    for (int kt = 0; kt < KT; ++kt) {
        __syncthreads();   // staging of buffer (kt&1) complete; prior reads of
                           // buffer ((kt+1)&1) finished -> safe to overwrite it
        const bf16* cur = &As[(kt & 1) * BUFSZ];

        // stage next k-step into the other buffer (overlaps with WMMA below)
        if (kt + 1 < KT)
            *(v4bf*)&As[((kt + 1) & 1) * BUFSZ + srow * LDSS + scol] =
                load4_cvt(ap + ((kt + 1) << 5));

        // B fragment: one contiguous 32B read per lane; prefetch next k-step
        v16bf b = *(const v16bf*)bp;
        if (kt + 1 < KT)
            __builtin_prefetch(bp + bstep, 0, 3);    // WGP-scope prefetch

        // two A fragments (rows 0-15 and 16-31): two ds_load_b128 each
        v8bf a0l = *(const v8bf*)&cur[arow * LDSS + aoff0];
        v8bf a0h = *(const v8bf*)&cur[arow * LDSS + aoff1];
        v8bf a1l = *(const v8bf*)&cur[(16 + arow) * LDSS + aoff0];
        v8bf a1h = *(const v8bf*)&cur[(16 + arow) * LDSS + aoff1];
        v16bf a0 = __builtin_shufflevector(a0l, a0h,
                    0, 1, 2, 3, 4, 5, 6, 7, 8, 9, 10, 11, 12, 13, 14, 15);
        v16bf a1 = __builtin_shufflevector(a1l, a1h,
                    0, 1, 2, 3, 4, 5, 6, 7, 8, 9, 10, 11, 12, 13, 14, 15);

        acc0 = __builtin_amdgcn_wmma_f32_16x16x32_bf16(
                   false, a0, false, b, (short)0, acc0, false, false);
        acc1 = __builtin_amdgcn_wmma_f32_16x16x32_bf16(
                   false, a1, false, b, (short)0, acc1, false, false);

        bp += bstep;
    }

    // C layout: VGPR v -> M = v (+8 for lanes 16..31), N = lane%16
    float bv = bias[col];
#pragma unroll
    for (int v = 0; v < 8; ++v) {
        int   row0 = rbase + v + (hi ? 8 : 0);
        int   row1 = row0 + 16;
        float v0 = acc0[v] + bv;
        float v1 = acc1[v] + bv;
        if (RELU) {
            v0 = v0 > 0.0f ? v0 : 0.0f;
            v1 = v1 > 0.0f ? v1 : 0.0f;
        }
        if (row0 < N_NODES) C[(size_t)row0 * N + col] = (TOUT)v0;
        if (row1 < N_NODES) C[(size_t)row1 * N + col] = (TOUT)v1;
    }
}

// ---------------------------------------------------------------------------
__global__ __launch_bounds__(256) void zero_f32_kernel(float4* __restrict__ p, int n4) {
    int i = blockIdx.x * 256 + threadIdx.x;
    if (i < n4) p[i] = make_float4(0.f, 0.f, 0.f, 0.f);
}

// gather h[src] (bf16), scatter-add into h_agg[dst] (f32 atomics).
// one thread per (edge, 4-feature chunk): 64 threads/edge, coalesced 8B gathers.
__global__ __launch_bounds__(256) void edge_scatter_kernel(const long long* __restrict__ ei,
                                                           const bf16* __restrict__ h,
                                                           float* __restrict__ hagg) {
    long long idx = (long long)blockIdx.x * 256 + threadIdx.x;
    int e  = (int)(idx >> 6);
    int c4 = (int)(idx & 63);
    if (e >= N_EDGES) return;
    int src = (int)ei[e];
    int dst = (int)ei[N_EDGES + e];
    v4bf hv = *(const v4bf*)&h[(size_t)src * EM_DIM + (c4 << 2)];
    float* out = &hagg[(size_t)dst * EM_DIM + (c4 << 2)];
#pragma unroll
    for (int i = 0; i < 4; ++i)
        atomicAdd(out + i, (float)hv[i]);
}

// ---------------------------------------------------------------------------
extern "C" void kernel_launch(void* const* d_in, const int* in_sizes, int n_in,
                              void* d_out, int out_size, void* d_ws, size_t ws_size,
                              hipStream_t stream) {
    const float*     x   = (const float*)d_in[0];
    const long long* ei  = (const long long*)d_in[1];
    const float*     W1i = (const float*)d_in[2];
    const float*     b1i = (const float*)d_in[3];
    const float*     W2i = (const float*)d_in[4];
    const float*     b2i = (const float*)d_in[5];
    const float*     W1f = (const float*)d_in[6];
    const float*     b1f = (const float*)d_in[7];
    const float*     W2f = (const float*)d_in[8];
    const float*     b2f = (const float*)d_in[9];
    float*           out = (float*)d_out;

    // ---- workspace carve-up (256B aligned) ----
    char*  ws  = (char*)d_ws;
    size_t off = 0;
    auto carve = [&](size_t bytes) {
        void* p = ws + off;
        off += (bytes + 255) & ~(size_t)255;
        return p;
    };
    const size_t PK1 = (size_t)(IN_DIM / 32) * (HI_DIM / 16) * 32 * 16 * 2;   // 128 KB
    const size_t PK2 = (size_t)(HI_DIM / 32) * (EM_DIM / 16) * 32 * 16 * 2;   // 256 KB
    const size_t PK3 = (size_t)(EM_DIM / 32) * (HI_DIM / 16) * 32 * 16 * 2;   // 256 KB
    const size_t PK4 = (size_t)(HI_DIM / 32) * (OUT_DIM / 16) * 32 * 16 * 2;  // 128 KB
    bf16*  pk1  = (bf16*)carve(PK1);
    bf16*  pk2  = (bf16*)carve(PK2);
    bf16*  pk3  = (bf16*)carve(PK3);
    bf16*  pk4  = (bf16*)carve(PK4);
    bf16*  h1   = (bf16*)carve((size_t)N_NODES * HI_DIM * 2);   // 51.2 MB (reused as h2)
    bf16*  h    = (bf16*)carve((size_t)N_NODES * EM_DIM * 2);   // 25.6 MB
    float* hagg = (float*)carve((size_t)N_NODES * EM_DIM * 4);  // 51.2 MB
    bf16*  h2   = h1;                                           // reuse

    // ---- pack weights into WMMA B-fragment layout (bf16) ----
    auto packBlocks = [](int K, int N) { return ((K >> 5) * (N >> 4) * 32 + 255) / 256; };
    pack_w_kernel<<<packBlocks(IN_DIM, HI_DIM), 256, 0, stream>>>(W1i, pk1, IN_DIM, HI_DIM);
    pack_w_kernel<<<packBlocks(HI_DIM, EM_DIM), 256, 0, stream>>>(W2i, pk2, HI_DIM, EM_DIM);
    pack_w_kernel<<<packBlocks(EM_DIM, HI_DIM), 256, 0, stream>>>(W1f, pk3, EM_DIM, HI_DIM);
    pack_w_kernel<<<packBlocks(HI_DIM, OUT_DIM), 256, 0, stream>>>(W2f, pk4, HI_DIM, OUT_DIM);

    const int MT = (N_NODES + 31) / 32;   // 1563 row tiles (last one partial)

    // f_init: Linear -> ReLU -> Linear
    gemm_bias_act<float, bf16, true,  IN_DIM, HI_DIM>
        <<<dim3(HI_DIM / 128, MT), 256, 0, stream>>>(x, pk1, b1i, h1);
    gemm_bias_act<bf16,  bf16, false, HI_DIM, EM_DIM>
        <<<dim3(EM_DIM / 128, MT), 256, 0, stream>>>(h1, pk2, b2i, h);

    // aggregation: zero accumulator, then gather/scatter-add over edges
    int n4 = N_NODES * EM_DIM / 4;
    zero_f32_kernel<<<(n4 + 255) / 256, 256, 0, stream>>>((float4*)hagg, n4);
    long long escatter_threads = (long long)N_EDGES * 64;
    int escatter_blocks = (int)((escatter_threads + 255) / 256);
    edge_scatter_kernel<<<escatter_blocks, 256, 0, stream>>>(ei, h, hagg);

    // f_final: Linear -> ReLU -> Linear
    gemm_bias_act<float, bf16, true,  EM_DIM, HI_DIM>
        <<<dim3(HI_DIM / 128, MT), 256, 0, stream>>>(hagg, pk3, b1f, h2);
    gemm_bias_act<bf16,  float, false, HI_DIM, OUT_DIM>
        <<<dim3(OUT_DIM / 128, MT), 256, 0, stream>>>(h2, pk4, b2f, out);

    (void)in_sizes; (void)n_in; (void)out_size; (void)ws_size;
}
